// GCN2_62371515072691
// MI455X (gfx1250) — matrix operand order repaired and checked by reading, not compiled, not run on hardware
//
#include <hip/hip_runtime.h>
#include <hip/hip_bf16.h>
#include <math.h>

// ---------------------------------------------------------------------------
// GCN on MI455X (gfx1250): f16 WMMA "TN" GEMMs (both operands K-major) with
// f32 accumulation, double-buffered LDS fed by CDNA5 async LDS loads
// (global_load_async_to_lds_b128 / ASYNCcnt) so tile fetch overlaps WMMA.
// Steady-state K loop is branch-free (unrolled x2, statically addressed bufs).
// Shapes: B=256, N=256 nodes, S=128, F=64 (63 real + 1 pad), H=1024, FCH=512
// ---------------------------------------------------------------------------

typedef _Float16 f16;
typedef __attribute__((ext_vector_type(16))) _Float16 v16h;
typedef __attribute__((ext_vector_type(8)))  float    v8f;

#define BATCH 256
#define NODES 256
#define SEQL  128
#define XROW  65      // x last-dim: 63 feats | station id | extra
#define FPAD  64      // padded feature dim (63 -> 64)

// ------------------------- small helper kernels ----------------------------

__global__ void zero_f16_kernel(f16* __restrict__ p, long long n) {
  long long i = (long long)blockIdx.x * blockDim.x + threadIdx.x;
  if (i < n) p[i] = (f16)0.0f;
}

// dst[n*Kp + k] = (k < Ksrc) ? src[k*N + n] : 0   (f32 -> f16, transposed+pad)
__global__ void cvt_t_kernel(const float* __restrict__ src, f16* __restrict__ dst,
                             int N, int Kp, int Ksrc) {
  int i = blockIdx.x * 256 + threadIdx.x;
  if (i >= N * Kp) return;
  int n = i / Kp, k = i % Kp;
  dst[i] = (k < Ksrc) ? (f16)src[(long long)k * N + n] : (f16)0.0f;
}

// dinv[n] = 1/sqrt(1 + sum_j adj[n,j])   (self-loop included)
__global__ void degree_kernel(const float* __restrict__ adj, float* __restrict__ dinv) {
  int n = blockIdx.x * 32 + threadIdx.x;   // grid 8 x 32
  float s = 1.0f;
  for (int j = 0; j < NODES; ++j) s += adj[n * NODES + j];
  dinv[n] = rsqrtf(s);
}

// An[i,j] = dinv[i] * (adj[i,j] + (i==j)) * dinv[j], f16 (symmetric)
__global__ void norm_adj_kernel(const float* __restrict__ adj, const float* __restrict__ dinv,
                                f16* __restrict__ Anh) {
  int idx = blockIdx.x * 256 + threadIdx.x;   // grid 256 x 256
  int i = idx >> 8, j = idx & 255;
  float a = adj[idx] + (i == j ? 1.0f : 0.0f);
  Anh[idx] = (f16)(dinv[i] * a * dinv[j]);
}

// Feat-major scatter: X0t[b][f][sid(b,j)] = x[b,j,f]  (X0t pre-zeroed)
__global__ void scatter_kernel(const float* __restrict__ x, f16* __restrict__ X0t) {
  int bj = blockIdx.x;                 // B*S blocks, 64 threads
  int b = bj >> 7, j = bj & 127;
  const float* row = x + (long long)b * (SEQL * XROW) + j * XROW;
  int sid = (int)row[63];
  int f = threadIdx.x;
  if (f < 63)
    X0t[(long long)b * (FPAD * NODES) + (long long)f * NODES + sid] = (f16)row[f];
}

// T[b,c] = sum_n An[g_sid(b), n] * Z2[c][b*256+n]  (both reads contiguous)
__global__ void gather_aggregate_kernel(const float* __restrict__ x, const f16* __restrict__ An,
                                        const f16* __restrict__ Z2, f16* __restrict__ T) {
  int b = blockIdx.x;
  int c = blockIdx.y * 128 + threadIdx.x;      // grid (256,4) x 128 -> 512 cols
  int gs = (int)x[(long long)b * (SEQL * XROW) + 127 * XROW + 63];
  const f16* an = An + (long long)gs * NODES;
  const f16* z  = Z2 + (long long)c * (BATCH * NODES) + b * NODES;
  float s = 0.0f;
  for (int n = 0; n < NODES; ++n)
    s += (float)an[n] * (float)z[n];
  T[b * 512 + c] = (f16)s;
}

// BatchNorm (training stats, biased var) + LeakyReLU on H[512][256] (feat-major)
__global__ void bn_leaky_kernel(float* __restrict__ H, const float* __restrict__ gamma,
                                const float* __restrict__ beta) {
  int c = blockIdx.x;            // 512 features
  int r = threadIdx.x;           // 256 batch
  float v = H[(long long)c * BATCH + r];       // contiguous
  __shared__ float s1[256], s2[256];
  s1[r] = v; s2[r] = v * v;
  __syncthreads();
  for (int off = 128; off > 0; off >>= 1) {
    if (r < off) { s1[r] += s1[r + off]; s2[r] += s2[r + off]; }
    __syncthreads();
  }
  float mean = s1[0] * (1.0f / 256.0f);
  float var  = s2[0] * (1.0f / 256.0f) - mean * mean;
  float hn = (v - mean) * rsqrtf(var + 1e-5f) * gamma[c] + beta[c];
  H[(long long)c * BATCH + r] = hn >= 0.0f ? hn : 0.01f * hn;
}

// out[b] = sigmoid(sum_c H[c][b]*fcW2[c] + fcb2)
__global__ void head_kernel(const float* __restrict__ H, const float* __restrict__ fcW2,
                            const float* __restrict__ fcb2, float* __restrict__ out) {
  int b = blockIdx.x;
  int t = threadIdx.x;           // 128
  float s = 0.0f;
  for (int c = t; c < 512; c += 128) s += H[(long long)c * BATCH + b] * fcW2[c];
  __shared__ float red[128];
  red[t] = s; __syncthreads();
  for (int off = 64; off > 0; off >>= 1) {
    if (t < off) red[t] += red[t + off];
    __syncthreads();
  }
  if (t == 0) out[b] = 1.0f / (1.0f + expf(-(red[0] + fcb2[0])));
}

// ------------------------- TN WMMA GEMM ------------------------------------
// C[n*ldc + m] = act(sum_k A[m*lda+k] * B[n*ldb+k] + bias)
// A: M x K (K-major), B: N x K (K-major), C: N x M row-major.
// Block tile 128x128, 8 waves (wave32), wave tile 32(M) x 64(N) = 2x4 WMMAs.
// K stepped by 32; double-buffered LDS via async LDS loads (ASYNCcnt).
// Requires M%128==0, N%128==0, K%64==0 (even step count).
// BIASMODE: 0=none, 1=bias[n], 2=bias[m].

#define LSTRIDE 40   // 80B LDS rows: 16B-aligned, conflict-friendly

template<int BIASMODE, bool RELU, bool OUTF16>
__global__ __launch_bounds__(256)
void gemm_tn_wmma_kernel(const f16* __restrict__ A, const f16* __restrict__ B,
                         void* __restrict__ Cp, const float* __restrict__ bias,
                         int K, int lda, int ldb, long long ldc,
                         long long aBatch, long long bBatch, long long cBatch)
{
  __shared__ f16 As[2][128 * LSTRIDE];
  __shared__ f16 Bs[2][128 * LSTRIDE];

  const int bz = blockIdx.z;
  A += bz * aBatch;
  B += bz * bBatch;
  f16*   Ch = (f16*)Cp   + bz * cBatch;
  float* Cf = (float*)Cp + bz * cBatch;

  const int tid  = threadIdx.x;
  const int lane = tid & 31;
  const int wave = tid >> 5;
  const int wm = (wave >> 1) * 32;        // wave sub-tile origin (M)
  const int wn = (wave & 1)  * 64;        // wave sub-tile origin (N)
  const long long tileM = (long long)blockIdx.y * 128;
  const long long tileN = (long long)blockIdx.x * 128;

  // cooperative staging: 128 rows x 32 halfs per tile; 2 threads per row,
  // each thread owns one 32B chunk (2 x b128, same LDS/global relative offset)
  const int lrow = tid >> 1, lcol = (tid & 1) * 16;
  const f16* aptr = A + (tileM + lrow) * lda + lcol;
  const f16* bptr = B + (tileN + lrow) * ldb + lcol;
  const unsigned aoff0 = (unsigned)(uintptr_t)&As[0][lrow * LSTRIDE + lcol];
  const unsigned aoff1 = (unsigned)(uintptr_t)&As[1][lrow * LSTRIDE + lcol];
  const unsigned boff0 = (unsigned)(uintptr_t)&Bs[0][lrow * LSTRIDE + lcol];
  const unsigned boff1 = (unsigned)(uintptr_t)&Bs[1][lrow * LSTRIDE + lcol];

  // per-lane fragment bases (k = 16*(lane/16) + e for both A and B; all K-contiguous)
  const int fl   = lane & 15;
  const int akb  = (lane >> 4) * 16;      // A: e0-15 at akb
  const int bkb  = (lane >> 4) * 16;      // B: e0-15 at bkb

  v8f acc[2][4] = {};

#define ASYNC_TILE(AO, BO)                                                     \
  do {                                                                         \
    asm volatile("global_load_async_to_lds_b128 %0, %2, off\n\t"               \
                 "global_load_async_to_lds_b128 %0, %2, off offset:16\n\t"     \
                 "global_load_async_to_lds_b128 %1, %3, off\n\t"               \
                 "global_load_async_to_lds_b128 %1, %3, off offset:16"         \
                 :: "v"(AO), "v"(BO), "v"(aptr), "v"(bptr) : "memory");        \
    aptr += 32; bptr += 32;                                                    \
  } while (0)

  // one K-step of math on a statically-addressed buffer
  auto compute = [&](const f16* Asc, const f16* Bsc) {
    v16h af[2], bf[4];
#pragma unroll
    for (int mi = 0; mi < 2; ++mi) {
      const f16* p = &Asc[(wm + mi * 16 + fl) * LSTRIDE + akb];
#pragma unroll
      for (int e = 0; e < 8; ++e) { af[mi][e] = p[e]; af[mi][8 + e] = p[8 + e]; }
    }
#pragma unroll
    for (int ni = 0; ni < 4; ++ni) {
      const f16* p = &Bsc[(wn + ni * 16 + fl) * LSTRIDE + bkb];
#pragma unroll
      for (int e = 0; e < 16; ++e) bf[ni][e] = p[e];
    }
#pragma unroll
    for (int mi = 0; mi < 2; ++mi)
#pragma unroll
      for (int ni = 0; ni < 4; ++ni)
        acc[mi][ni] = __builtin_amdgcn_wmma_f32_16x16x32_f16(
            false, af[mi], false, bf[ni], (short)0, acc[mi][ni], false, false);
  };

  // prologue: fetch tile 0 into buffer 0
  ASYNC_TILE(aoff0, boff0);

  // steady state: branch-free, two K-steps per trip (nsteps is always even)
  const int nsteps = K >> 5;
  for (int i = 0; i + 2 < nsteps; i += 2) {
    ASYNC_TILE(aoff1, boff1);
    asm volatile("s_wait_asynccnt 0x4" ::: "memory");
    __syncthreads();
    compute(As[0], Bs[0]);
    __syncthreads();

    ASYNC_TILE(aoff0, boff0);
    asm volatile("s_wait_asynccnt 0x4" ::: "memory");
    __syncthreads();
    compute(As[1], Bs[1]);
    __syncthreads();
  }
  // tail: last two K-steps
  ASYNC_TILE(aoff1, boff1);
  asm volatile("s_wait_asynccnt 0x4" ::: "memory");
  __syncthreads();
  compute(As[0], Bs[0]);
  __syncthreads();
  asm volatile("s_wait_asynccnt 0x0" ::: "memory");
  __syncthreads();
  compute(As[1], Bs[1]);
#undef ASYNC_TILE

  // epilogue: per acc, 8 contiguous m per lane -> one 16B (f16) / 32B (f32) store
#pragma unroll
  for (int mi = 0; mi < 2; ++mi) {
#pragma unroll
    for (int ni = 0; ni < 4; ++ni) {
      long long n  = tileN + wn + ni * 16 + fl;
      long long m0 = tileM + wm + mi * 16 + ((lane >> 4) << 3);
      float bn_ = (BIASMODE == 1) ? bias[n] : 0.0f;
      if (OUTF16) {
        f16 h8[8];
#pragma unroll
        for (int v = 0; v < 8; ++v) {
          float val = acc[mi][ni][v];
          if (BIASMODE == 1) val += bn_;
          if (BIASMODE == 2) val += bias[m0 + v];
          if (RELU) val = val > 0.0f ? val : 0.0f;
          h8[v] = (f16)val;
        }
        *(uint4*)(&Ch[n * ldc + m0]) = *(uint4*)h8;
      } else {
        float f8[8];
#pragma unroll
        for (int v = 0; v < 8; ++v) {
          float val = acc[mi][ni][v];
          if (BIASMODE == 1) val += bn_;
          if (BIASMODE == 2) val += bias[m0 + v];
          if (RELU) val = val > 0.0f ? val : 0.0f;
          f8[v] = val;
        }
        *(float4*)(&Cf[n * ldc + m0])     = ((float4*)f8)[0];
        *(float4*)(&Cf[n * ldc + m0 + 4]) = ((float4*)f8)[1];
      }
    }
  }
}

// ------------------------------ host side ----------------------------------

extern "C" void kernel_launch(void* const* d_in, const int* in_sizes, int n_in,
                              void* d_out, int out_size, void* d_ws, size_t ws_size,
                              hipStream_t stream) {
  const float* x     = (const float*)d_in[0];
  const float* adj   = (const float*)d_in[1];
  const float* W1    = (const float*)d_in[2];
  const float* b1    = (const float*)d_in[3];
  const float* W2    = (const float*)d_in[4];
  const float* b2    = (const float*)d_in[5];
  const float* W3    = (const float*)d_in[6];
  const float* b3    = (const float*)d_in[7];
  const float* fcW1  = (const float*)d_in[8];
  const float* fcb1  = (const float*)d_in[9];
  const float* gamma = (const float*)d_in[10];
  const float* beta  = (const float*)d_in[11];
  const float* fcW2  = (const float*)d_in[12];
  const float* fcb2  = (const float*)d_in[13];
  float* out = (float*)d_out;

  char* wsb = (char*)d_ws;
  size_t off = 0;
  auto alloc = [&](size_t bytes) -> void* {
    off = (off + 255) & ~(size_t)255;
    void* p = wsb + off;
    off += bytes;
    return p;
  };

  float* dinv  = (float*)alloc(NODES * sizeof(float));
  f16*   Anh   = (f16*)alloc((size_t)NODES * NODES * 2);          // 128 KB
  f16*   W1t   = (f16*)alloc((size_t)512 * FPAD * 2);             // [512][64]
  f16*   W2t   = (f16*)alloc((size_t)512 * 512 * 2);              // [512][512]
  f16*   W3t   = (f16*)alloc((size_t)1024 * 512 * 2);             // [1024][512]
  f16*   fcW1t = (f16*)alloc((size_t)512 * 1024 * 2);             // [512][1024]
  f16*   X0t   = (f16*)alloc((size_t)BATCH * FPAD * NODES * 2);   // 8 MB  [b][64][256]
  f16*   Y1    = (f16*)alloc((size_t)NODES * BATCH * FPAD * 2);   // 8 MB  [node][b*64+f]
  f16*   Z1    = (f16*)alloc((size_t)512 * BATCH * NODES * 2);    // 64 MB [512][b*256+node]
  f16*   Y2    = (f16*)alloc((size_t)BATCH * NODES * 512 * 2);    // 64 MB [b][node][512]
  f16*   Z2    = Z1;   // alias: Z1 dead once Y2 = An(.)Z1 is done
  f16*   T     = (f16*)alloc((size_t)BATCH * 512 * 2);            // [256][512]
  f16*   G     = (f16*)alloc((size_t)BATCH * 1024 * 2);           // [256][1024]
  float* H     = (float*)alloc((size_t)512 * BATCH * 4);          // [512][256] f32
  // peak workspace ~148 MB

  // --- prep
  {
    long long nX0 = (long long)BATCH * FPAD * NODES;
    zero_f16_kernel<<<(unsigned)((nX0 + 255) / 256), 256, 0, stream>>>(X0t, nX0);
  }
  degree_kernel<<<8, 32, 0, stream>>>(adj, dinv);
  norm_adj_kernel<<<NODES * NODES / 256, 256, 0, stream>>>(adj, dinv, Anh);
  scatter_kernel<<<BATCH * SEQL, 64, 0, stream>>>(x, X0t);
  cvt_t_kernel<<<512 * FPAD / 256, 256, 0, stream>>>(W1, W1t, 512, FPAD, 63);
  cvt_t_kernel<<<512 * 512 / 256, 256, 0, stream>>>(W2, W2t, 512, 512, 512);
  cvt_t_kernel<<<1024 * 512 / 256, 256, 0, stream>>>(W3, W3t, 1024, 512, 512);
  cvt_t_kernel<<<512 * 1024 / 256, 256, 0, stream>>>(fcW1, fcW1t, 512, 1024, 1024);

  // --- Op1: Y1[node][b*64+f] = sum_n X0t[(b,f)][n] * An[node][n]
  //     (all batches flatten: M=16384 vs shared An)
  gemm_tn_wmma_kernel<0, false, true><<<dim3(NODES / 128, BATCH * FPAD / 128, 1), 256, 0, stream>>>(
      X0t, Anh, (void*)Y1, nullptr, NODES, NODES, NODES, (long long)BATCH * FPAD,
      0LL, 0LL, 0LL);

  // --- Op2 (batched): Z1[f2][b*256+node] = relu(sum_f Y1[node][b*64+f]*W1t[f2][f] + b1)
  gemm_tn_wmma_kernel<1, true, true><<<dim3(512 / 128, NODES / 128, BATCH), 256, 0, stream>>>(
      Y1, W1t, (void*)Z1, b1, FPAD, BATCH * FPAD, FPAD, (long long)BATCH * NODES,
      (long long)FPAD, 0LL, (long long)NODES);

  // --- Op3 (batched): Y2[b][node'][f2] = sum_n Z1[f2][b*256+n] * An[node'][n]
  gemm_tn_wmma_kernel<0, false, true><<<dim3(NODES / 128, 512 / 128, BATCH), 256, 0, stream>>>(
      Z1, Anh, (void*)Y2, nullptr, NODES, BATCH * NODES, NODES, 512LL,
      (long long)NODES, 0LL, (long long)NODES * 512);

  // --- Op4: Z2[f3][b*256+node] = relu(sum_f2 Y2[(b,node)][f2]*W2t[f3][f2] + b2)
  gemm_tn_wmma_kernel<1, true, true><<<dim3(512 / 128, BATCH * NODES / 128, 1), 256, 0, stream>>>(
      Y2, W2t, (void*)Z2, b2, 512, 512, 512, (long long)BATCH * NODES,
      0LL, 0LL, 0LL);

  // --- layer 3, gathered rows only
  gather_aggregate_kernel<<<dim3(BATCH, 4), 128, 0, stream>>>(x, Anh, Z2, T);

  // --- Op5: G[b][f] = relu(sum_k W3t[f][k]*T[b][k] + b3[f])   (bias over m)
  gemm_tn_wmma_kernel<2, true, true><<<dim3(BATCH / 128, 1024 / 128, 1), 256, 0, stream>>>(
      W3t, T, (void*)G, b3, 512, 512, 512, 1024LL, 0LL, 0LL, 0LL);

  // --- Op6: H[f][b] = sum_k G[b][k]*fcW1t[f][k] + fcb1[f]   (f32 out, feat-major)
  gemm_tn_wmma_kernel<1, false, false><<<dim3(512 / 128, BATCH / 128, 1), 256, 0, stream>>>(
      G, fcW1t, (void*)H, fcb1, 1024, 1024, 1024, (long long)BATCH, 0LL, 0LL, 0LL);

  // --- BN (training stats) + LeakyReLU, then dot + sigmoid
  bn_leaky_kernel<<<512, 256, 0, stream>>>(H, gamma, beta);
  head_kernel<<<BATCH, 128, 0, stream>>>(H, fcW2, fcb2, out);

  (void)in_sizes; (void)n_in; (void)out_size; (void)ws_size;
}
